// MultiHeadSelfAttentionLayer_14894946583182
// MI455X (gfx1250) — compile-verified, matrix operand order checked by reading, not run on hardware
//
#include <hip/hip_runtime.h>
#include <hip/hip_bf16.h>

// MI455X / gfx1250 multi-head self-attention:
//   f16 WMMA (f32 accumulate) for all four matmuls, flash-style streaming
//   softmax, double-buffered GEMM k-loops, async global->LDS staging.

typedef __attribute__((ext_vector_type(16))) _Float16 v16h;
typedef __attribute__((ext_vector_type(8)))  _Float16 v8h;
typedef __attribute__((ext_vector_type(8)))  float    v8f;

#define B_   8
#define N_   1024
#define DIN  768
#define H_   12
#define DH   64
#define DOUT 768
#define MROWS (B_ * N_)   // 8192

#if __has_builtin(__builtin_amdgcn_global_load_async_to_lds_b128) && \
    __has_builtin(__builtin_amdgcn_s_wait_asynccnt)
#define HAVE_ASYNC_LDS 1
#endif

#ifdef HAVE_ASYNC_LDS
// Builtin signature (from hipcc diagnostic): arg0 is a non-const pointer to
// a native int __vector(4) in AS1 (global); arg1 is the LDS-side pointer.
typedef int v4i_nat __attribute__((vector_size(16)));
typedef v4i_nat __attribute__((address_space(1))) v4i_g;
typedef v4i_nat __attribute__((address_space(3))) v4i_l;
#endif

__device__ __forceinline__ v8f vzero() {
  v8f z = {0.f, 0.f, 0.f, 0.f, 0.f, 0.f, 0.f, 0.f};
  return z;
}

// D = A*B + C, 16x16x32 f16 -> f32
__device__ __forceinline__ v8f wmma32(v16h a, v16h b, v8f c) {
  return __builtin_amdgcn_wmma_f32_16x16x32_f16(
      /*neg_a=*/false, a, /*neg_b=*/false, b,
      /*c_mod=*/(short)0, c, /*reuse_a=*/false, /*reuse_b=*/false);
}

// Load a 16(x)x32(k) f16 tile in the CDNA5 WMMA A/B register layout from a
// row-major array `p` with leading dimension `ld` (rows are the 16-dim).
//  lane L (L<16): row = row0+L, holds k = {0..7, 16..23}+k0
//  lane L+16    : row = row0+L, holds k = {8..15, 24..31}+k0
__device__ __forceinline__ v16h load_tile16x32(const _Float16* p, int ld,
                                               int row0, int k0, int lane) {
  const int r = lane & 15, hi = lane >> 4;
  const _Float16* base = p + (size_t)(row0 + r) * (size_t)ld + (k0 + hi * 8);
  v8h lo = *(const v8h*)(base);        // k = hi*8 .. hi*8+7
  v8h hf = *(const v8h*)(base + 16);   // k = 16+hi*8 .. 16+hi*8+7
  v16h out;
#pragma unroll
  for (int i = 0; i < 8; ++i) { out[i] = lo[i]; out[i + 8] = hf[i]; }
  return out;
}

// 16-byte global -> LDS copy: async (ASYNCcnt-tracked, no VGPR bounce) when
// the toolchain exposes the gfx1250 builtin, else a plain int4 bounce.
__device__ __forceinline__ void stage16(const void* g, void* l) {
#ifdef HAVE_ASYNC_LDS
  __builtin_amdgcn_global_load_async_to_lds_b128((v4i_g*)g, (v4i_l*)l, 0, 0);
#else
  *(int4*)l = *(const int4*)g;
#endif
}

__device__ __forceinline__ void stage_wait() {
#ifdef HAVE_ASYNC_LDS
  __builtin_amdgcn_s_wait_asynccnt(0);
#endif
}

// ---------------------------------------------------------------- convert
__global__ void cvt_f32_f16(const float* __restrict__ s,
                            _Float16* __restrict__ d, int n) {
  int i = blockIdx.x * blockDim.x + threadIdx.x;
  if (i < n) d[i] = (_Float16)s[i];
}

// ------------------------------------------------------------- GEMM core
// One wave computes 32 rows x 64 cols of C[m,n] = sum_k A[m,k]*W[n,k].
// Double-buffered k-loop: tiles for k+32 are loaded while k's 8 WMMAs run,
// so s_wait_loadcnt covers a full k-step of latency instead of zero.
#define GEMM_BODY(AP, WP, LDK)                                               \
  v8f acc0 = vzero(), acc1 = vzero(), acc2 = vzero(), acc3 = vzero();        \
  v8f acc4 = vzero(), acc5 = vzero(), acc6 = vzero(), acc7 = vzero();        \
  v16h a0 = load_tile16x32(AP, LDK, m0, 0, lane);                            \
  v16h a1 = load_tile16x32(AP, LDK, m0 + 16, 0, lane);                       \
  v16h b0 = load_tile16x32(WP, LDK, c0 + 0,  0, lane);                       \
  v16h b1 = load_tile16x32(WP, LDK, c0 + 16, 0, lane);                       \
  v16h b2 = load_tile16x32(WP, LDK, c0 + 32, 0, lane);                       \
  v16h b3 = load_tile16x32(WP, LDK, c0 + 48, 0, lane);                       \
  for (int k0 = 0; k0 < (LDK) - 32; k0 += 32) {                              \
    const int kn = k0 + 32;                                                  \
    v16h na0 = load_tile16x32(AP, LDK, m0, kn, lane);                        \
    v16h na1 = load_tile16x32(AP, LDK, m0 + 16, kn, lane);                   \
    v16h nb0 = load_tile16x32(WP, LDK, c0 + 0,  kn, lane);                   \
    v16h nb1 = load_tile16x32(WP, LDK, c0 + 16, kn, lane);                   \
    v16h nb2 = load_tile16x32(WP, LDK, c0 + 32, kn, lane);                   \
    v16h nb3 = load_tile16x32(WP, LDK, c0 + 48, kn, lane);                   \
    acc0 = wmma32(a0, b0, acc0); acc1 = wmma32(a1, b0, acc1);                \
    acc2 = wmma32(a0, b1, acc2); acc3 = wmma32(a1, b1, acc3);                \
    acc4 = wmma32(a0, b2, acc4); acc5 = wmma32(a1, b2, acc5);                \
    acc6 = wmma32(a0, b3, acc6); acc7 = wmma32(a1, b3, acc7);                \
    a0 = na0; a1 = na1; b0 = nb0; b1 = nb1; b2 = nb2; b3 = nb3;              \
  }                                                                          \
  acc0 = wmma32(a0, b0, acc0); acc1 = wmma32(a1, b0, acc1);                  \
  acc2 = wmma32(a0, b1, acc2); acc3 = wmma32(a1, b1, acc3);                  \
  acc4 = wmma32(a0, b2, acc4); acc5 = wmma32(a1, b2, acc5);                  \
  acc6 = wmma32(a0, b3, acc6); acc7 = wmma32(a1, b3, acc7);

// ---------------------------------------------------------------- QKV GEMM
// q[b,h,n,e] = sum_d x[b,n,d] * W[h,e,d]; 64-col block == one full head.
// grid = (MROWS/32, 3*H_), block = 32
__global__ void __launch_bounds__(32)
qkv_gemm(const _Float16* __restrict__ xh,
         const _Float16* __restrict__ Wqh, const _Float16* __restrict__ Wkh,
         const _Float16* __restrict__ Wvh,
         _Float16* __restrict__ Qh, _Float16* __restrict__ Kh,
         _Float16* __restrict__ Vh) {
  const int lane = threadIdx.x & 31;
  const int m0 = blockIdx.x * 32;
  const int cb = blockIdx.y;           // 0..35
  const int mat = cb / H_;             // 0=Q 1=K 2=V
  const int h   = cb % H_;
  const int c0  = h * DH;
  const _Float16* W = (mat == 0) ? Wqh : (mat == 1) ? Wkh : Wvh;
  _Float16* dst     = (mat == 0) ? Qh  : (mat == 1) ? Kh  : Vh;

  GEMM_BODY(xh, W, DIN)

  // C layout: vgpr i, lanes 0-15 -> row i col lane; lanes 16-31 -> row i+8.
  const int b  = m0 >> 10;        // 32-row tile never crosses batch boundary
  const int n0 = m0 & (N_ - 1);
  const int hi = lane >> 4, col = lane & 15;
  _Float16* out = dst + (size_t)(b * H_ + h) * N_ * DH;
#pragma unroll
  for (int i = 0; i < 8; ++i) {
    const size_t r0 = (size_t)(n0 + i + 8 * hi) * DH;        // rows m0..+15
    const size_t r1 = (size_t)(n0 + 16 + i + 8 * hi) * DH;   // rows m0+16..
    out[r0 + 0  + col] = (_Float16)acc0[i];
    out[r1 + 0  + col] = (_Float16)acc1[i];
    out[r0 + 16 + col] = (_Float16)acc2[i];
    out[r1 + 16 + col] = (_Float16)acc3[i];
    out[r0 + 32 + col] = (_Float16)acc4[i];
    out[r1 + 32 + col] = (_Float16)acc5[i];
    out[r0 + 48 + col] = (_Float16)acc6[i];
    out[r1 + 48 + col] = (_Float16)acc7[i];
  }
}

// ---------------------------------------------------------------- attention
// grid = (N_/64, B_*H_), block = 128 (4 waves). Wave w owns 16 query rows.
// K and V chunks staged into LDS with async b128 copies (ASYNCcnt), V then
// transposed LDS->LDS so V tiles read as WMMA-B-layout ds_load_b128.
__global__ void __launch_bounds__(128)
attention_kernel(const _Float16* __restrict__ Qh,
                 const _Float16* __restrict__ Kh,
                 const _Float16* __restrict__ Vh,
                 _Float16* __restrict__ Aout) {
  __shared__ alignas(16) _Float16 Kbuf[32 * DH];     // [key][d]
  __shared__ alignas(16) _Float16 Vn[32 * DH];       // [key][d] (natural)
  __shared__ alignas(16) _Float16 Vt[DH * 32];       // [d][key] (transposed)
  __shared__ alignas(16) _Float16 Pbuf[4][16 * 32];  // per-wave [qrow][key]

  const int t = threadIdx.x;
  const int lane = t & 31, w = t >> 5;
  const int hi = lane >> 4, col = lane & 15;
  const int bh = blockIdx.y;
  const int b = bh / H_, h = bh % H_;
  const int q0 = blockIdx.x * 64 + w * 16;

  const _Float16* Qg = Qh + (size_t)bh * N_ * DH;
  const _Float16* Kg = Kh + (size_t)bh * N_ * DH;
  const _Float16* Vg = Vh + (size_t)bh * N_ * DH;
  _Float16* Pw = &Pbuf[w][0];

  // Q tile (16 x 64) loaded once, as two A-layout tiles.
  v16h qa0 = load_tile16x32(Qg, DH, q0, 0, lane);
  v16h qa1 = load_tile16x32(Qg, DH, q0, 32, lane);

  v8f O0 = vzero(), O1 = vzero(), O2 = vzero(), O3 = vzero();
  float m[8], l[8];
#pragma unroll
  for (int i = 0; i < 8; ++i) { m[i] = -1e30f; l[i] = 0.f; }

  for (int j = 0; j < N_; j += 32) {
    // --- async-stage K and V chunks (each 4 KB contiguous) -------------
    {
      const char* gk = (const char*)(Kg + (size_t)j * DH);
      const char* gv = (const char*)(Vg + (size_t)j * DH);
      char* lk = (char*)Kbuf;
      char* lv = (char*)Vn;
      stage16(gk + t * 16,        lk + t * 16);
      stage16(gk + 2048 + t * 16, lk + 2048 + t * 16);
      stage16(gv + t * 16,        lv + t * 16);
      stage16(gv + 2048 + t * 16, lv + 2048 + t * 16);
      stage_wait();
    }
    __syncthreads();
    // V transpose, LDS -> LDS (16 f16 per thread)
#pragma unroll
    for (int e = 0; e < 16; ++e) {
      int idx = t * 16 + e;
      int key = idx >> 6, d = idx & 63;
      Vt[d * 32 + key] = Vn[key * DH + d];
    }
    __syncthreads();

    // --- S = Q K^T for 2 key tiles (B-layout straight from Kbuf) -------
    v16h kb00 = load_tile16x32(Kbuf, DH, 0,  0,  lane);
    v16h kb01 = load_tile16x32(Kbuf, DH, 0,  32, lane);
    v16h kb10 = load_tile16x32(Kbuf, DH, 16, 0,  lane);
    v16h kb11 = load_tile16x32(Kbuf, DH, 16, 32, lane);
    v8f S0 = vzero(), S1 = vzero();
    S0 = wmma32(qa0, kb00, S0);
    S0 = wmma32(qa1, kb01, S0);
    S1 = wmma32(qa0, kb10, S1);
    S1 = wmma32(qa1, kb11, S1);

    // --- online softmax (rows live per (vgpr, lane-half); xor masks 1..8
    //     reduce across the 16 lanes of each half) ----------------------
    const float sc = 0.125f;  // 1/sqrt(64)
#pragma unroll
    for (int i = 0; i < 8; ++i) {
      float s0 = S0[i] * sc, s1 = S1[i] * sc;
      float rm = fmaxf(s0, s1);
#pragma unroll
      for (int msk = 1; msk < 16; msk <<= 1)
        rm = fmaxf(rm, __shfl_xor(rm, msk, 32));
      float mn = fmaxf(m[i], rm);
      float alpha = __expf(m[i] - mn);
      float p0 = __expf(s0 - mn);
      float p1 = __expf(s1 - mn);
      float rs = p0 + p1;
#pragma unroll
      for (int msk = 1; msk < 16; msk <<= 1)
        rs += __shfl_xor(rs, msk, 32);
      l[i] = l[i] * alpha + rs;
      m[i] = mn;
      O0[i] *= alpha; O1[i] *= alpha; O2[i] *= alpha; O3[i] *= alpha;
      const int row = i + 8 * hi;              // C-layout row for this lane
      Pw[row * 32 + col]      = (_Float16)p0;  // keys j .. j+15
      Pw[row * 32 + 16 + col] = (_Float16)p1;  // keys j+16 .. j+31
    }
    __syncthreads();  // P C-layout -> A-layout bounce needs cross-lane sync

    // --- O += P (16x32) * V (32x64), 4 B-tiles from transposed Vt ------
    v16h pa  = load_tile16x32(Pw, 32, 0,  0, lane);
    v16h vb0 = load_tile16x32(Vt, 32, 0,  0, lane);
    v16h vb1 = load_tile16x32(Vt, 32, 16, 0, lane);
    v16h vb2 = load_tile16x32(Vt, 32, 32, 0, lane);
    v16h vb3 = load_tile16x32(Vt, 32, 48, 0, lane);
    O0 = wmma32(pa, vb0, O0);
    O1 = wmma32(pa, vb1, O1);
    O2 = wmma32(pa, vb2, O2);
    O3 = wmma32(pa, vb3, O3);
    __syncthreads();  // protect Kbuf/Vn/Vt before next chunk overwrites
  }

  // --- normalize and write concat-head activation [B*N, 768] f16 -------
#pragma unroll
  for (int i = 0; i < 8; ++i) {
    float inv = 1.0f / l[i];
    int q = q0 + i + 8 * hi;
    _Float16* dst = Aout + (size_t)(b * N_ + q) * DOUT + h * DH;
    dst[0  + col] = (_Float16)(O0[i] * inv);
    dst[16 + col] = (_Float16)(O1[i] * inv);
    dst[32 + col] = (_Float16)(O2[i] * inv);
    dst[48 + col] = (_Float16)(O3[i] * inv);
  }
}

// ---------------------------------------------------------------- out proj
// y[m,o] = sum_c A[m,c]*Wo[o,c] + bo[o]; fp32 out. grid=(256,12), block=32
__global__ void __launch_bounds__(32)
out_proj_gemm(const _Float16* __restrict__ A, const _Float16* __restrict__ Woh,
              const float* __restrict__ bo, float* __restrict__ y) {
  const int lane = threadIdx.x & 31;
  const int m0 = blockIdx.x * 32;
  const int c0 = blockIdx.y * 64;

  GEMM_BODY(A, Woh, DOUT)

  const int hi = lane >> 4, col = lane & 15;
#pragma unroll
  for (int i = 0; i < 8; ++i) {
    const int r0 = m0 + i + 8 * hi;
    const int r1 = r0 + 16;
    float bz0 = bo[c0 + 0  + col], bz1 = bo[c0 + 16 + col];
    float bz2 = bo[c0 + 32 + col], bz3 = bo[c0 + 48 + col];
    float* d0 = y + (size_t)r0 * DOUT + c0;
    float* d1 = y + (size_t)r1 * DOUT + c0;
    d0[0  + col] = acc0[i] + bz0;
    d1[0  + col] = acc1[i] + bz0;
    d0[16 + col] = acc2[i] + bz1;
    d1[16 + col] = acc3[i] + bz1;
    d0[32 + col] = acc4[i] + bz2;
    d1[32 + col] = acc5[i] + bz2;
    d0[48 + col] = acc6[i] + bz3;
    d1[48 + col] = acc7[i] + bz3;
  }
}

// ---------------------------------------------------------------- launcher
extern "C" void kernel_launch(void* const* d_in, const int* in_sizes, int n_in,
                              void* d_out, int out_size, void* d_ws,
                              size_t ws_size, hipStream_t stream) {
  (void)in_sizes; (void)n_in; (void)out_size; (void)ws_size;
  const float* x  = (const float*)d_in[0];
  const float* Wq = (const float*)d_in[1];
  const float* Wk = (const float*)d_in[2];
  const float* Wv = (const float*)d_in[3];
  const float* Wo = (const float*)d_in[4];
  const float* bo = (const float*)d_in[5];

  // workspace carve-up (f16 elements); ~55 MB total.
  _Float16* ws = (_Float16*)d_ws;
  const size_t XN = (size_t)MROWS * DIN;        // 6,291,456
  const size_t WN = (size_t)DOUT * DIN;         //   589,824
  const size_t QN = (size_t)B_ * H_ * N_ * DH;  // 6,291,456
  _Float16* xh  = ws;
  _Float16* Wqh = xh + XN;
  _Float16* Wkh = Wqh + WN;
  _Float16* Wvh = Wkh + WN;
  _Float16* Woh = Wvh + WN;
  _Float16* Qh  = Woh + WN;
  _Float16* Kh  = Qh + QN;
  _Float16* Vh  = Kh + QN;
  _Float16* Aout = xh;  // alias: xh is dead after qkv_gemm, Aout written after

  const int CT = 256;
  cvt_f32_f16<<<(int)((XN + CT - 1) / CT), CT, 0, stream>>>(x, xh, (int)XN);
  cvt_f32_f16<<<(int)((WN + CT - 1) / CT), CT, 0, stream>>>(Wq, Wqh, (int)WN);
  cvt_f32_f16<<<(int)((WN + CT - 1) / CT), CT, 0, stream>>>(Wk, Wkh, (int)WN);
  cvt_f32_f16<<<(int)((WN + CT - 1) / CT), CT, 0, stream>>>(Wv, Wvh, (int)WN);
  cvt_f32_f16<<<(int)((WN + CT - 1) / CT), CT, 0, stream>>>(Wo, Woh, (int)WN);

  dim3 g1(MROWS / 32, 3 * H_);
  qkv_gemm<<<g1, 32, 0, stream>>>(xh, Wqh, Wkh, Wvh, Qh, Kh, Vh);

  dim3 g2(N_ / 64, B_ * H_);
  attention_kernel<<<g2, 128, 0, stream>>>(Qh, Kh, Vh, Aout);

  dim3 g3(MROWS / 32, DOUT / 64);
  out_proj_gemm<<<g3, 32, 0, stream>>>(Aout, Woh, bo, (float*)d_out);
}
